// Decoder_44573170598295
// MI455X (gfx1250) — matrix operand order repaired and checked by reading, not compiled, not run on hardware
//
#include <hip/hip_runtime.h>

// ---------------------------------------------------------------------------
// Problem constants (from the reference)
// ---------------------------------------------------------------------------
constexpr int L   = 6;
constexpr int B   = 2;
constexpr int F   = 16;
constexpr int T   = 64;
constexpr int S   = F * T;      // 1024
constexpr int E   = 768;
constexpr int H   = 12;
constexpr int DK  = 64;
constexpr int HD  = H * DK;     // 768
constexpr int DFF = 3072;
constexpr int BS  = B * S;      // 2048

// GEMM block tile: 128x64 per block (8 waves, 32x32 per wave), K-step 32
constexpr int BM = 128;
constexpr int BN = 64;
constexpr int BK = 32;
constexpr int LDA_S = 20;  // dwords per A row in LDS (= 40 bf16, padded from 32)
constexpr int LDB_S = 20;  // dwords per B row (n-major) in LDS

typedef unsigned short u16;
typedef __attribute__((ext_vector_type(16))) __bf16 v16bf;
typedef __attribute__((ext_vector_type(2)))  __bf16 v2bf;
typedef __attribute__((ext_vector_type(8)))  float  v8f;
typedef __attribute__((ext_vector_type(4)))  int    v4i;

// ---------------------------------------------------------------------------
// fp32 -> bf16 (round to nearest even)
// ---------------------------------------------------------------------------
__device__ __forceinline__ u16 f2bf(float f) {
    unsigned int u = __builtin_bit_cast(unsigned int, f);
    u += 0x7fffu + ((u >> 16) & 1u);
    return (u16)(u >> 16);
}
__device__ __forceinline__ unsigned int pack_bf16x2(float lo, float hi) {
    unsigned int ulo = __builtin_bit_cast(unsigned int, lo);
    unsigned int uhi = __builtin_bit_cast(unsigned int, hi);
    ulo += 0x7fffu + ((ulo >> 16) & 1u);
    uhi += 0x7fffu + ((uhi >> 16) & 1u);
    return (ulo >> 16) | (uhi & 0xffff0000u);
}

// ---------------------------------------------------------------------------
// Async global->LDS 16B copy (CDNA5 GLOBAL_LOAD_ASYNC_TO_LDS_B128, ASYNCcnt),
// with a vectorized synchronous fallback if the builtins are absent.
// ---------------------------------------------------------------------------
#if defined(__has_builtin)
#  if __has_builtin(__builtin_amdgcn_global_load_async_to_lds_b128) && \
      __has_builtin(__builtin_amdgcn_s_wait_asynccnt)
#    define USE_ASYNC_COPY 1
#  endif
#endif

#ifdef USE_ASYNC_COPY
typedef __attribute__((address_space(1))) v4i* gptr_v4i;
typedef __attribute__((address_space(3))) v4i* lptr_v4i;
#endif

__device__ __forceinline__ void copy16_g2l(const void* g, void* l) {
#ifdef USE_ASYNC_COPY
    gptr_v4i gp = (gptr_v4i)(unsigned long long)g;
    lptr_v4i lp = (lptr_v4i)(unsigned int)(unsigned long long)l;
    __builtin_amdgcn_global_load_async_to_lds_b128(gp, lp, 0, 0);
#else
    *(uint4*)l = *(const uint4*)g;
#endif
}
// wait for ALL outstanding async copies
__device__ __forceinline__ void copy_fence_all() {
#ifdef USE_ASYNC_COPY
    __builtin_amdgcn_s_wait_asynccnt(0);
#endif
}
// wait until only the 3 most recent copies (next tile) remain in flight;
// ASYNCcnt retires in order, so this guarantees the current tile has landed
__device__ __forceinline__ void copy_fence_keep3() {
#ifdef USE_ASYNC_COPY
    __builtin_amdgcn_s_wait_asynccnt(3);
#endif
}

// ---------------------------------------------------------------------------
// NT bf16 WMMA GEMM:  C[z] = alpha * A[z](MxK) * B[z]^T           (opt. ReLU)
//   A: bf16 [M][K] row-major (lda).  B: bf16 [N][K] row-major (ldb).
//   OUT==0: C fp32 [M][N] (ldc)
//   OUT==1: C bf16 [M][N] (ldc)
//   OUT==2: C bf16 written per-head transposed: vT[b][h][d][s]  (QKV V path;
//           M = b*S+s fused, N = h*DK+d; 8 consecutive s per lane -> b128 store)
//   Batch z = zo*zdiv + zi with independent outer/inner strides (head slicing).
// Double-buffered LDS: async copies of tile t+1 overlap the WMMAs on tile t.
// All dims in this problem are exact multiples of the tiles: no edge checks.
// ---------------------------------------------------------------------------
template <int OUT>
__global__ __launch_bounds__(256) void gemm_nt_bf16(
    const u16* __restrict__ A, size_t sAo, size_t sAi, int lda,
    const u16* __restrict__ Bm, size_t sBo, size_t sBi, int ldb,
    void* __restrict__ Cv, size_t sCo, size_t sCi, int ldc,
    int K, int zdiv, float alpha, int relu)
{
    __shared__ unsigned int Asu[2][BM * LDA_S];
    __shared__ unsigned int Bsu[2][BN * LDB_S];

    const int tid  = threadIdx.x;
    const int lane = tid & 31;
    const int wave = tid >> 5;
    const int wm   = wave >> 1;      // 0..3
    const int wn   = wave & 1;       // 0..1
    const int g    = lane >> 4;      // lane group (0/1)
    const int l15  = lane & 15;

    const int row0 = blockIdx.y * BM;
    const int col0 = blockIdx.x * BN;
    const int zo = blockIdx.z / zdiv;
    const int zi = blockIdx.z % zdiv;
    A  += (size_t)zo * sAo + (size_t)zi * sAi;
    Bm += (size_t)zo * sBo + (size_t)zi * sBi;

    // stage one 128x32 A tile (2 chunks/thread) + 64x32 B tile (1 chunk/thread)
    auto stage = [&](int k0, int bufi) {
        const u16* Abase = A + (size_t)row0 * lda + k0;
#pragma unroll
        for (int i = 0; i < 2; ++i) {
            const int idx = tid + i * 256;
            const int r = idx >> 2, c = idx & 3;
            copy16_g2l(Abase + (size_t)r * lda + c * 8,
                       &Asu[bufi][r * LDA_S + c * 4]);
        }
        const u16* Bbase = Bm + (size_t)col0 * ldb + k0;
        const int n = tid >> 2, c = tid & 3;
        copy16_g2l(Bbase + (size_t)n * ldb + c * 8,
                   &Bsu[bufi][n * LDB_S + c * 4]);
    };

    v8f acc[2][2] = {};

    stage(0, 0);
    int buf = 0;
    for (int k0 = 0; k0 < K; k0 += BK, buf ^= 1) {
        if (k0 + BK < K) {
            // issue next tile into the other buffer, then wait only for the
            // current tile's 3 chunks (in-order retirement)
            stage(k0 + BK, buf ^ 1);
            copy_fence_keep3();
        } else {
            copy_fence_all();
        }
        __syncthreads();

        // prefetch hint two tiles ahead (global_prefetch_b8)
        if (k0 + 2 * BK < K) {
            __builtin_prefetch(A  + (size_t)(row0 + (tid >> 2)) * lda + k0 + 2 * BK, 0, 1);
            __builtin_prefetch(Bm + (size_t)(col0 + (tid >> 2)) * ldb + k0 + 2 * BK, 0, 1);
        }

        // ---- build fragments per ISA 16-bit layouts, issue 4 WMMAs ----
        v16bf af[2], bfr[2];
#pragma unroll
        for (int tm = 0; tm < 2; ++tm) {
            const int m = wm * 32 + tm * 16 + l15;
#pragma unroll
            for (int i = 0; i < 8; ++i) {
                // A 16x32: VGPR i<4 -> K = g*8 + 2i(+1); i>=4 -> 16 + g*8 + 2(i-4)(+1)
                const int kk = ((i >> 2) << 4) + g * 8 + ((i & 3) << 1);
                v2bf p = __builtin_bit_cast(v2bf, Asu[buf][m * LDA_S + (kk >> 1)]);
                af[tm][2 * i]     = p[0];
                af[tm][2 * i + 1] = p[1];
            }
        }
#pragma unroll
        for (int tn = 0; tn < 2; ++tn) {
            const int n = wn * 32 + tn * 16 + l15;
#pragma unroll
            for (int i = 0; i < 8; ++i) {
                // B 32x16: VGPR i<4 -> K = g*8 + 2i(+1); i>=4 -> 16 + g*8 + 2(i-4)(+1)
                const int kk = ((i >> 2) << 4) + g * 8 + ((i & 3) << 1);
                v2bf p = __builtin_bit_cast(v2bf, Bsu[buf][n * LDB_S + (kk >> 1)]);
                bfr[tn][2 * i]     = p[0];
                bfr[tn][2 * i + 1] = p[1];
            }
        }
#pragma unroll
        for (int tm = 0; tm < 2; ++tm)
#pragma unroll
            for (int tn = 0; tn < 2; ++tn)
                acc[tm][tn] = __builtin_amdgcn_wmma_f32_16x16x32_bf16(
                    false, af[tm], false, bfr[tn], (short)0, acc[tm][tn],
                    false, false);
        __syncthreads();   // all waves done reading buf before it is re-staged
    }

    // ---- epilogue ----
#pragma unroll
    for (int tm = 0; tm < 2; ++tm) {
#pragma unroll
        for (int tn = 0; tn < 2; ++tn) {
            const int n = col0 + wn * 32 + tn * 16 + l15;
            float vv[8];
#pragma unroll
            for (int r = 0; r < 8; ++r) {
                float t = acc[tm][tn][r] * alpha;
                vv[r] = relu ? fmaxf(t, 0.0f) : t;
            }
            if (OUT == 0) {
                float* C = (float*)Cv + zo * sCo + zi * sCi;
#pragma unroll
                for (int r = 0; r < 8; ++r) {
                    const int m = row0 + wm * 32 + tm * 16 + g * 8 + r;
                    C[(size_t)m * ldc + n] = vv[r];
                }
            } else if (OUT == 1) {
                u16* C = (u16*)Cv + zo * sCo + zi * sCi;
#pragma unroll
                for (int r = 0; r < 8; ++r) {
                    const int m = row0 + wm * 32 + tm * 16 + g * 8 + r;
                    C[(size_t)m * ldc + n] = f2bf(vv[r]);
                }
            } else {
                // vT[b][h][d][s]: rows m..m+7 are 8 consecutive s -> one b128
                const int m0 = row0 + wm * 32 + tm * 16 + g * 8;
                const int b  = m0 >> 10;             // S == 1024
                const int s  = m0 & (S - 1);
                const size_t base =
                    ((size_t)(b * H + (n >> 6)) * DK + (n & (DK - 1))) * S + s;
                uint4 u = make_uint4(pack_bf16x2(vv[0], vv[1]),
                                     pack_bf16x2(vv[2], vv[3]),
                                     pack_bf16x2(vv[4], vv[5]),
                                     pack_bf16x2(vv[6], vv[7]));
                *(uint4*)((u16*)Cv + base) = u;
            }
        }
    }
}

// ---------------------------------------------------------------------------
// Weight transpose+convert: in fp32 [K][N] -> out bf16 [N][K], LDS-tiled 32x32
// ---------------------------------------------------------------------------
__global__ __launch_bounds__(256) void wtrans_kernel(
    const float* __restrict__ in, u16* __restrict__ out, int K, int N)
{
    __shared__ float t[32][33];
    const int n0 = blockIdx.x * 32, k0 = blockIdx.y * 32;
    const int tx = threadIdx.x & 31, ty = threadIdx.x >> 5;
#pragma unroll
    for (int j = 0; j < 4; ++j)
        t[ty + 8 * j][tx] = in[(size_t)(k0 + ty + 8 * j) * N + n0 + tx];
    __syncthreads();
#pragma unroll
    for (int j = 0; j < 4; ++j)
        out[(size_t)(n0 + ty + 8 * j) * K + k0 + tx] = f2bf(t[tx][ty + 8 * j]);
}

// ---------------------------------------------------------------------------
// x = x_in + t_pos[frame] + s_pos[token]  (fp32 + bf16 shadow)
// ---------------------------------------------------------------------------
__global__ __launch_bounds__(256) void pos_add_kernel(
    const float* __restrict__ xin, const float* __restrict__ tpos,
    const float* __restrict__ spos, float* __restrict__ xo,
    u16* __restrict__ xb)
{
    const int idx = blockIdx.x * 256 + threadIdx.x;      // BS*E total
    const int e = idx % E;
    const int s = (idx / E) % S;
    const float v = xin[idx] + tpos[(s / T) * E + e] + spos[(s % T) * E + e];
    xo[idx] = v;
    xb[idx] = f2bf(v);
}

// pad flag per (b,s): first embed channel == 0 (post positional add)
__global__ __launch_bounds__(256) void pad_flags_kernel(
    const float* __restrict__ x, float* __restrict__ padf)
{
    const int i = blockIdx.x * 256 + threadIdx.x;        // BS total
    padf[i] = (x[(size_t)i * E] == 0.0f) ? 1.0f : 0.0f;
}

// ---------------------------------------------------------------------------
// causal+pad mask then row softmax, in place on [B*H,S,S] fp32 probs (d_out),
// plus a bf16 shadow copy for the PV WMMA GEMM.
// grid: (S, B*H), 256 threads, 4 columns per thread
// ---------------------------------------------------------------------------
__global__ __launch_bounds__(256) void softmax_mask_kernel(
    float* __restrict__ attn, const float* __restrict__ padf,
    u16* __restrict__ pb)
{
    const int q  = blockIdx.x;
    const int bh = blockIdx.y;
    const int b  = bh / H;
    float* row = attn + (size_t)bh * S * S + (size_t)q * S;
    u16*  prow = pb   + (size_t)bh * S * S + (size_t)q * S;
    const float* pr = padf + (size_t)b * S;
    const int tid = threadIdx.x;

    float v[4];
    float mx = -3.4e38f;
#pragma unroll
    for (int i = 0; i < 4; ++i) {
        const int c = tid + i * 256;
        float sv = row[c];
        const bool masked = (c > q) || (pr[c] != 0.0f);
        sv = masked ? -1e9f : sv;
        v[i] = sv;
        mx = fmaxf(mx, sv);
    }
    __shared__ float red[256];
    red[tid] = mx; __syncthreads();
    for (int o = 128; o > 0; o >>= 1) {
        if (tid < o) red[tid] = fmaxf(red[tid], red[tid + o]);
        __syncthreads();
    }
    mx = red[0]; __syncthreads();

    float sum = 0.0f;
#pragma unroll
    for (int i = 0; i < 4; ++i) { v[i] = __expf(v[i] - mx); sum += v[i]; }
    red[tid] = sum; __syncthreads();
    for (int o = 128; o > 0; o >>= 1) {
        if (tid < o) red[tid] += red[tid + o];
        __syncthreads();
    }
    const float inv = 1.0f / red[0];
#pragma unroll
    for (int i = 0; i < 4; ++i) {
        const int c = tid + i * 256;
        const float p = v[i] * inv;
        row[c]  = p;
        prow[c] = f2bf(p);
    }
}

// ---------------------------------------------------------------------------
// x = LayerNorm(a + x)*gamma + beta   (fp32 spine + bf16 shadow, row = E=768)
// ---------------------------------------------------------------------------
__global__ __launch_bounds__(256) void add_ln_kernel(
    const float* __restrict__ a, float* __restrict__ x, u16* __restrict__ xb,
    const float* __restrict__ gm, const float* __restrict__ bt)
{
    const int row = blockIdx.x;                          // BS rows
    const size_t base = (size_t)row * E;
    const int tid = threadIdx.x;

    float r[3];
    float s = 0.0f, ss = 0.0f;
#pragma unroll
    for (int i = 0; i < 3; ++i) {
        const int e = tid + i * 256;
        const float t = a[base + e] + x[base + e];
        r[i] = t; s += t; ss += t * t;
    }
    __shared__ float rs[256], rq[256];
    rs[tid] = s; rq[tid] = ss; __syncthreads();
    for (int o = 128; o > 0; o >>= 1) {
        if (tid < o) { rs[tid] += rs[tid + o]; rq[tid] += rq[tid + o]; }
        __syncthreads();
    }
    const float mean = rs[0] * (1.0f / E);
    const float var  = rq[0] * (1.0f / E) - mean * mean;
    const float inv  = rsqrtf(var + 1e-5f);
#pragma unroll
    for (int i = 0; i < 3; ++i) {
        const int e = tid + i * 256;
        const float o = (r[i] - mean) * inv * gm[e] + bt[e];
        x[base + e]  = o;
        xb[base + e] = f2bf(o);
    }
}

__global__ __launch_bounds__(256) void copy_kernel(
    const float* __restrict__ src, float* __restrict__ dst)
{
    const int i = blockIdx.x * 256 + threadIdx.x;
    dst[i] = src[i];
}

// ---------------------------------------------------------------------------
// Host-side orchestration
// ---------------------------------------------------------------------------
extern "C" void kernel_launch(void* const* d_in, const int* in_sizes, int n_in,
                              void* d_out, int out_size, void* d_ws, size_t ws_size,
                              hipStream_t stream)
{
    const float* x_in  = (const float*)d_in[0];
    const float* t_pos = (const float*)d_in[1];
    const float* s_pos = (const float*)d_in[2];
    const float* Wq    = (const float*)d_in[3];
    const float* Wk    = (const float*)d_in[4];
    const float* Wv    = (const float*)d_in[5];
    const float* Wo    = (const float*)d_in[6];
    const float* ln1g  = (const float*)d_in[7];
    const float* ln1b  = (const float*)d_in[8];
    const float* W1    = (const float*)d_in[9];
    const float* W2    = (const float*)d_in[10];
    const float* ln2g  = (const float*)d_in[11];
    const float* ln2b  = (const float*)d_in[12];

    float* out      = (float*)d_out;
    float* out_x    = out;                               // [B,S,E]
    float* out_attn = out + (size_t)BS * E;              // [L,B,H,S,S]

    // workspace layout: fp32 spine, then bf16 region
    float* xw  = (float*)d_ws;                           // [BS,E]
    float* tmp = xw  + (size_t)BS * E;                   // [BS,E]
    float* pad = tmp + (size_t)BS * E;                   // [BS]
    u16* xb   = (u16*)(pad + BS);                        // [BS,E]
    u16* qb   = xb   + (size_t)BS * E;                   // [BS,HD]
    u16* kb   = qb   + (size_t)BS * HD;                  // [BS,HD]
    u16* vT   = kb   + (size_t)BS * HD;                  // [B,H,DK,S]
    u16* ctxb = vT   + (size_t)BS * HD;                  // [BS,HD]
    u16* h1b  = ctxb + (size_t)BS * HD;                  // [BS,DFF]
    u16* pb   = h1b  + (size_t)BS * DFF;                 // [B*H,S,S]
    u16* wqT  = pb   + (size_t)B * H * S * S;            // per-layer scratch:
    u16* wkT  = wqT  + (size_t)E * HD;                   //   [HD,E]
    u16* wvT  = wkT  + (size_t)E * HD;
    u16* woT  = wvT  + (size_t)E * HD;                   //   [E,HD]
    u16* w1T  = woT  + (size_t)HD * E;                   //   [DFF,E]
    u16* w2T  = w1T  + (size_t)E * DFF;                  //   [E,DFF]

    const float scale = 0.125f;                          // 1/sqrt(64)

    pos_add_kernel<<<(BS * E) / 256, 256, 0, stream>>>(x_in, t_pos, s_pos, xw, xb);
    pad_flags_kernel<<<BS / 256, 256, 0, stream>>>(xw, pad);

    const dim3 gProj(HD / BN, BS / BM, 1);               // (12,16,1)
    const dim3 gScore(S / BN, S / BM, B * H);            // (16, 8,24)
    const dim3 gCtx(DK / BN, S / BM, B * H);             // ( 1, 8,24)
    const dim3 gFfn1(DFF / BN, BS / BM, 1);              // (48,16,1)
    const dim3 gFfn2(E / BN, BS / BM, 1);                // (12,16,1)

    for (int l = 0; l < L; ++l) {
        float* attn_l = out_attn + (size_t)l * B * H * S * S;

        // weights -> bf16, transposed to [N][K] (NT layout), per-layer scratch
        wtrans_kernel<<<dim3(HD / 32, E / 32), 256, 0, stream>>>(
            Wq + (size_t)l * E * HD, wqT, E, HD);
        wtrans_kernel<<<dim3(HD / 32, E / 32), 256, 0, stream>>>(
            Wk + (size_t)l * E * HD, wkT, E, HD);
        wtrans_kernel<<<dim3(HD / 32, E / 32), 256, 0, stream>>>(
            Wv + (size_t)l * E * HD, wvT, E, HD);
        wtrans_kernel<<<dim3(E / 32, HD / 32), 256, 0, stream>>>(
            Wo + (size_t)l * HD * E, woT, HD, E);
        wtrans_kernel<<<dim3(DFF / 32, E / 32), 256, 0, stream>>>(
            W1 + (size_t)l * E * DFF, w1T, E, DFF);
        wtrans_kernel<<<dim3(E / 32, DFF / 32), 256, 0, stream>>>(
            W2 + (size_t)l * DFF * E, w2T, DFF, E);

        // Q,K (bf16 out) and V (bf16 out, pre-transposed per head)
        gemm_nt_bf16<1><<<gProj, 256, 0, stream>>>(
            xb, 0, 0, E,  wqT, 0, 0, E,  qb, 0, 0, HD,  E, 1, 1.0f, 0);
        gemm_nt_bf16<1><<<gProj, 256, 0, stream>>>(
            xb, 0, 0, E,  wkT, 0, 0, E,  kb, 0, 0, HD,  E, 1, 1.0f, 0);
        gemm_nt_bf16<2><<<gProj, 256, 0, stream>>>(
            xb, 0, 0, E,  wvT, 0, 0, E,  vT, 0, 0, HD,  E, 1, 1.0f, 0);

        // scores[b,h] = scale * Q[b,h] @ K[b,h]^T  -> fp32 directly in d_out
        gemm_nt_bf16<0><<<gScore, 256, 0, stream>>>(
            qb, (size_t)S * HD, DK, HD,
            kb, (size_t)S * HD, DK, HD,
            attn_l, (size_t)H * S * S, (size_t)S * S, S,
            DK, H, scale, 0);

        // mask + softmax in place; also emit bf16 probs for the PV GEMM
        softmax_mask_kernel<<<dim3(S, B * H), 256, 0, stream>>>(attn_l, pad, pb);

        // ctx[b,h] = P[b,h] @ V[b,h]  (B operand = vT, already [d][s])
        gemm_nt_bf16<1><<<gCtx, 256, 0, stream>>>(
            pb, (size_t)H * S * S, (size_t)S * S, S,
            vT, (size_t)H * DK * S, (size_t)DK * S, S,
            ctxb, (size_t)S * HD, DK, HD,
            S, H, 1.0f, 0);

        // output projection + residual LayerNorm
        gemm_nt_bf16<0><<<gProj, 256, 0, stream>>>(
            ctxb, 0, 0, HD,  woT, 0, 0, HD,  tmp, 0, 0, E,  HD, 1, 1.0f, 0);
        add_ln_kernel<<<BS, 256, 0, stream>>>(tmp, xw, xb,
                                              ln1g + l * E, ln1b + l * E);

        // FFN: h1 = relu(x @ W1) (bf16), tmp = h1 @ W2 (fp32), x = LN(tmp + x)
        gemm_nt_bf16<1><<<gFfn1, 256, 0, stream>>>(
            xb, 0, 0, E,  w1T, 0, 0, E,  h1b, 0, 0, DFF,  E, 1, 1.0f, 1);
        gemm_nt_bf16<0><<<gFfn2, 256, 0, stream>>>(
            h1b, 0, 0, DFF,  w2T, 0, 0, DFF,  tmp, 0, 0, E,  DFF, 1, 1.0f, 0);
        add_ln_kernel<<<BS, 256, 0, stream>>>(tmp, xw, xb,
                                              ln2g + l * E, ln2b + l * E);
    }

    // final hidden states -> d_out[0 : B*S*E]
    copy_kernel<<<(BS * E) / 256, 256, 0, stream>>>(xw, out_x);
}
